// RFLayerwise_36756330120126
// MI455X (gfx1250) — matrix operand order repaired and checked by reading, not compiled, not run on hardware
//
#include <hip/hip_runtime.h>

typedef unsigned short ushort_t;
typedef unsigned int   uint_t;

typedef __attribute__((ext_vector_type(16))) __bf16 v16bf;
typedef __attribute__((ext_vector_type(8)))  float  v8f;
typedef int i32x4 __attribute__((vector_size(16)));   // matches builtin param

#if defined(__HIP_DEVICE_COMPILE__) && __has_builtin(__builtin_amdgcn_global_load_async_to_lds_b128)
#define USE_ASYNC_LDS 1
#else
#define USE_ASYNC_LDS 0
#endif

__device__ __forceinline__ ushort_t f32_to_bf16(float f) {
  uint_t u = __float_as_uint(f);
  u += 0x7fffu + ((u >> 16) & 1u);   // round-to-nearest-even
  return (ushort_t)(u >> 16);
}

__device__ __forceinline__ float bf16_to_f32(ushort_t h) {
  return __uint_as_float(((uint_t)h) << 16);
}

// 16B global -> LDS copy; async (ASYNCcnt) when available, sync fallback.
__device__ __forceinline__ void cp_b128(const ushort_t* g, ushort_t* l) {
#if USE_ASYNC_LDS
  __builtin_amdgcn_global_load_async_to_lds_b128((i32x4*)g, (i32x4*)l, 0, 0);
#else
  *(uint4*)l = *(const uint4*)g;
#endif
}

__device__ __forceinline__ void wait_async_done() {
#if USE_ASYNC_LDS
#if __has_builtin(__builtin_amdgcn_s_wait_asynccnt)
  __builtin_amdgcn_s_wait_asynccnt(0);
#else
  asm volatile("s_wait_asynccnt 0" ::: "memory");
#endif
#endif
}

// ---------------------------------------------------------------------------
// fp32 -> bf16 conversion (one-time for x and all weights)
// ---------------------------------------------------------------------------
__global__ void cvt_f32_to_bf16(const float* __restrict__ s,
                                ushort_t* __restrict__ d, int n) {
  int i = blockIdx.x * blockDim.x + threadIdx.x;
  if (i < n) d[i] = f32_to_bf16(s[i]);
}

// ---------------------------------------------------------------------------
// C[M,N] = A[M,K] * W[N,K]^T   (bf16 in memory, fp32 accumulation)
// Block: 256 threads (8 wave32) as a 4x2 wave grid; WG tile 128x128;
// each wave owns 32 rows x 64 cols (2 A-frags x 4 B-frags = 8 WMMA tiles).
// K-step 64 (two x32 chunks), double-buffered LDS via async global->LDS,
// explicit 2-deep B-fragment pipeline so ds_load latency hides under WMMA.
// ---------------------------------------------------------------------------
union AF { uint4 u[2]; v16bf v; };
union BF { uint4 u[2]; v16bf v; };

__global__ __launch_bounds__(256)
void gemm_bt_bf16(const ushort_t* __restrict__ A,   // [M,K] bf16 row-major
                  const ushort_t* __restrict__ W,   // [N,K] bf16 row-major
                  ushort_t* __restrict__ C,         // [M,N] bf16 row-major
                  int M, int N, int K, int relu)
{
  // [buf][mat][128 rows][64 cols] bf16 : 4 * 16 KB = 64 KB
  __shared__ ushort_t smem[2 * 2 * 128 * 64];
  const int TILE = 128 * 64;

  const int tid  = threadIdx.x;
  const int lane = tid & 31;
  const int wave = tid >> 5;          // 0..7
  const int l16  = lane & 15;
  const int hsel = lane >> 4;         // 0 or 1 (lane half)

  const int waveM = wave & 3;         // 4 row-groups of 32 rows
  const int waveN = wave >> 2;        // 2 col-groups of 64 cols

  const int tileM = blockIdx.y * 128;
  const int tileN = blockIdx.x * 128;

  v8f acc0[4] = {};                   // rows [waveM*32,   +16)
  v8f acc1[4] = {};                   // rows [waveM*32+16,+16)

  // staging: 2 threads per row, 32 bf16 (64B) each -> 4 x b128 per matrix
  const int ldRow = tid >> 1;                 // 0..127
  const int ldCol = (tid & 1) * 32;           // 0 or 32 (elements)
  const ushort_t* aG = A + (size_t)(tileM + ldRow) * K + ldCol;
  const ushort_t* wG = W + (size_t)(tileN + ldRow) * K + ldCol;
  ushort_t* aL[2] = { smem + 0 * TILE + ldRow * 64 + ldCol,
                      smem + 2 * TILE + ldRow * 64 + ldCol };
  ushort_t* wL[2] = { smem + 1 * TILE + ldRow * 64 + ldCol,
                      smem + 3 * TILE + ldRow * 64 + ldCol };

  // compute-side LDS element offsets inside a buffer
  const int aOff0 = (waveM * 32 + 0  + l16) * 64 + hsel * 8;   // A frag m=0
  const int aOff1 = (waveM * 32 + 16 + l16) * 64 + hsel * 8;   // A frag m=1
  const int wOffB = (waveN * 64 + l16) * 64 + hsel * 8;        // + n*16*64

  const int kSteps = K >> 6;          // K-step = 64

  // prologue: stage tile 0 into buffer 0
#pragma unroll
  for (int j = 0; j < 4; ++j) cp_b128(aG + j * 8, aL[0] + j * 8);
#pragma unroll
  for (int j = 0; j < 4; ++j) cp_b128(wG + j * 8, wL[0] + j * 8);

  for (int s = 0; s < kSteps; ++s) {
    const int b = s & 1;

    wait_async_done();        // my copies for tile s have landed in LDS
    __syncthreads();          // everyone's landed + everyone done with buf b^1

    if (s + 1 < kSteps) {     // stage tile s+1 into the other buffer (overlaps)
      const int nb = b ^ 1;
      const int kNext = (s + 1) << 6;
#pragma unroll
      for (int j = 0; j < 4; ++j) cp_b128(aG + kNext + j * 8, aL[nb] + j * 8);
#pragma unroll
      for (int j = 0; j < 4; ++j) cp_b128(wG + kNext + j * 8, wL[nb] + j * 8);
    }

    const ushort_t* Ab = smem + (2 * b + 0) * TILE;
    const ushort_t* Wb = smem + (2 * b + 1) * TILE;

#pragma unroll
    for (int kk = 0; kk < 64; kk += 32) {
      // A fragments (16x32, ISA 7.12.2): v[0..7]=K[kb..+7], v[8..15]=K[kb+16..]
      AF af0, af1;
      af0.u[0] = *(const uint4*)(Ab + aOff0 + kk);
      af0.u[1] = *(const uint4*)(Ab + aOff0 + kk + 16);
      af1.u[0] = *(const uint4*)(Ab + aOff1 + kk);
      af1.u[1] = *(const uint4*)(Ab + aOff1 + kk + 16);

      // 2-deep B-fragment pipeline: issue loads for n+1 before using n
      BF bf[2];
      {
        const ushort_t* bp = Wb + wOffB + kk;
        bf[0].u[0] = ((const uint4*)bp)[0];
        bf[0].u[1] = ((const uint4*)bp)[2];
      }
#pragma unroll
      for (int n = 0; n < 4; ++n) {
        if (n < 3) {
          const ushort_t* bp = Wb + wOffB + (n + 1) * (16 * 64) + kk;
          bf[(n + 1) & 1].u[0] = ((const uint4*)bp)[0];
          bf[(n + 1) & 1].u[1] = ((const uint4*)bp)[2];
        }
        acc0[n] = __builtin_amdgcn_wmma_f32_16x16x32_bf16(
            false, af0.v, false, bf[n & 1].v, (short)0, acc0[n], false, false);
        acc1[n] = __builtin_amdgcn_wmma_f32_16x16x32_bf16(
            false, af1.v, false, bf[n & 1].v, (short)0, acc1[n], false, false);
      }
    }
  }

  // Epilogue: C/D layout — VGPR j holds M=j (lanes 0-15) / M=j+8 (lanes 16-31)
  const size_t row0 = (size_t)(tileM + waveM * 32 + hsel * 8);
#pragma unroll
  for (int n = 0; n < 4; ++n) {
    const int c = tileN + waveN * 64 + n * 16 + l16;
#pragma unroll
    for (int j = 0; j < 8; ++j) {
      float v = acc0[n][j];
      if (relu) v = fmaxf(v, 0.0f);
      C[(row0 + j) * (size_t)N + c] = f32_to_bf16(v);
    }
#pragma unroll
    for (int j = 0; j < 8; ++j) {
      float v = acc1[n][j];
      if (relu) v = fmaxf(v, 0.0f);
      C[(row0 + 16 + j) * (size_t)N + c] = f32_to_bf16(v);
    }
  }
}

// ---------------------------------------------------------------------------
// out[b, o] = (1/H) * sum_k Y[b,k] * beta[k,o]    H=1024, OUT=10
// beta (40KB) staged in LDS; one thread per row; LDS reads are broadcasts.
// ---------------------------------------------------------------------------
__global__ __launch_bounds__(256)
void final_beta(const ushort_t* __restrict__ Y, const float* __restrict__ beta,
                float* __restrict__ out, int Bn)
{
  __shared__ float bs[1024 * 10];
  for (int i = threadIdx.x; i < 1024 * 10; i += 256) bs[i] = beta[i];
  __syncthreads();

  const int row = blockIdx.x * 256 + threadIdx.x;
  if (row >= Bn) return;

  float acc[10];
#pragma unroll
  for (int o = 0; o < 10; ++o) acc[o] = 0.0f;

  const ushort_t* yr = Y + (size_t)row * 1024;
  for (int k = 0; k < 1024; ++k) {
    const float v = bf16_to_f32(yr[k]);
#pragma unroll
    for (int o = 0; o < 10; ++o) acc[o] += v * bs[k * 10 + o];
  }
#pragma unroll
  for (int o = 0; o < 10; ++o)
    out[(size_t)row * 10 + o] = acc[o] * (1.0f / 1024.0f);
}

// ---------------------------------------------------------------------------
extern "C" void kernel_launch(void* const* d_in, const int* in_sizes, int n_in,
                              void* d_out, int out_size, void* d_ws, size_t ws_size,
                              hipStream_t stream) {
  const int B = 16384, D = 64, H = 1024;

  const float* x     = (const float*)d_in[0];
  const float* rp[4] = {(const float*)d_in[1], (const float*)d_in[3],
                        (const float*)d_in[5], (const float*)d_in[7]};
  const float* w[4]  = {(const float*)d_in[2], (const float*)d_in[4],
                        (const float*)d_in[6], (const float*)d_in[8]};
  const float* beta  = (const float*)d_in[9];
  float* out = (float*)d_out;

  // workspace carve-out (~80 MB): bf16 copies of x/weights + two activations
  char* p = (char*)d_ws;
  ushort_t* xb = (ushort_t*)p;            p += (size_t)B * D * 2;   // 2 MB
  ushort_t* rpb[4];
  rpb[0] = (ushort_t*)p;                  p += (size_t)H * D * 2;   // 128 KB
  for (int l = 1; l < 4; ++l) { rpb[l] = (ushort_t*)p; p += (size_t)H * H * 2; }
  ushort_t* wb[4];
  for (int l = 0; l < 4; ++l) { wb[l]  = (ushort_t*)p; p += (size_t)H * H * 2; }
  ushort_t* tbuf = (ushort_t*)p;          p += (size_t)B * H * 2;   // 32 MB
  ushort_t* ybuf = (ushort_t*)p;          p += (size_t)B * H * 2;   // 32 MB

  // one-time fp32 -> bf16 conversions
  cvt_f32_to_bf16<<<(B * D + 255) / 256, 256, 0, stream>>>(x, xb, B * D);
  cvt_f32_to_bf16<<<(H * D + 255) / 256, 256, 0, stream>>>(rp[0], rpb[0], H * D);
  for (int l = 1; l < 4; ++l)
    cvt_f32_to_bf16<<<(H * H + 255) / 256, 256, 0, stream>>>(rp[l], rpb[l], H * H);
  for (int l = 0; l < 4; ++l)
    cvt_f32_to_bf16<<<(H * H + 255) / 256, 256, 0, stream>>>(w[l], wb[l], H * H);

  const dim3 grid(H / 128, B / 128);  // (8, 128)
  const dim3 blk(256);

  // layer 0: t = x @ rp0^T (K=64); y = relu(t @ w0^T)
  gemm_bt_bf16<<<grid, blk, 0, stream>>>(xb,   rpb[0], tbuf, B, H, D, 0);
  gemm_bt_bf16<<<grid, blk, 0, stream>>>(tbuf, wb[0],  ybuf, B, H, H, 1);
  // layers 1..3
  for (int l = 1; l < 4; ++l) {
    gemm_bt_bf16<<<grid, blk, 0, stream>>>(ybuf, rpb[l], tbuf, B, H, H, 0);
    gemm_bt_bf16<<<grid, blk, 0, stream>>>(tbuf, wb[l],  ybuf, B, H, H, 1);
  }

  // final projection + scale
  final_beta<<<B / 256, 256, 0, stream>>>(ybuf, beta, out, B);
}
